// HeteGCNLayer_39883066310781
// MI455X (gfx1250) — matrix operand order repaired and checked by reading, not compiled, not run on hardware
//
#include <hip/hip_runtime.h>
#include <hip/hip_bf16.h>
#include <math.h>

// ---------------------------------------------------------------------------
// HeteGCN layer for MI455X (gfx1250, wave32, WMMA)
//   N=100000 nodes/type, DIN=256, DOUT=64, ATT=64, E=3.2M edges/relation
// Pipeline:
//   1) 4x dense GEMM [N,256]@[256,64] via v_wmma_f32_16x16x32_f16 (f32 accum)
//   2) zero scatter buffers
//   3) 2x SpMM scatter-add (L2-resident tables) with global_atomic_add_f32
//   4) fold attention mats: vk = w_keys@w_att[:64], vq = w_query@w_att[64:]
//   5) per-node ELU attention softmax(2) + blend + bias, wave32 reductions
// ---------------------------------------------------------------------------

typedef __attribute__((ext_vector_type(16))) _Float16 v16h;
typedef __attribute__((ext_vector_type(8)))  float    v8f;

#define NN   100000
#define DIN  256
#define DOUT 64

// ---------------------------------------------------------------------------
// GEMM: OUT[N,64] = X[N,256] @ W[256,64], f16 WMMA with f32 accumulate.
// Block = 256 threads = 8 waves; wave w handles 16-row tile (blockIdx*8+w).
// W is staged once per block into LDS, pre-swizzled into per-fragment layout:
//   fragment (kt in 0..7, nt in 0..3): 32 lanes x 16 halves, contiguous.
// B-fragment layout (16x16x32 f16): lanes 0-15 hold col N=lane, K=kt*32+{0..7, 16..23};
//   lanes 16-31 hold col N=lane-16, K=kt*32+{8..15, 24..31} (2 halves per VGPR).
// A-fragment layout: lanes 0-15: row M=lane, K = kt*32 + {0..7, 16..23};
//   lanes 16-31: row M=lane-16, K = kt*32 + {8..15, 24..31}.
// ---------------------------------------------------------------------------
__global__ __launch_bounds__(256) void gemm_wmma_256x64(
    const float* __restrict__ X, const float* __restrict__ W,
    float* __restrict__ OUT, int nrows)
{
    __shared__ __attribute__((aligned(32))) _Float16 wlds[8 * 4 * 32 * 16]; // 32 KB

    const int tid = threadIdx.x;

    // --- stage + swizzle W into LDS: thread t owns W row k=t (64 f32) ---
    {
        const int k   = tid;
        const int kt  = k >> 5;
        const int kin = k & 31;
        const int idx = (kin & 7) + ((kin & 16) >> 1);
        const int lsel = (kin & 8) ? 16 : 0;
        const float* wrow = W + k * DOUT;
        #pragma unroll 8
        for (int n = 0; n < DOUT; ++n) {
            const int nt   = n >> 4;
            const int lane = lsel + (n & 15);
            wlds[(((kt * 4 + nt) * 32 + lane) << 4) + idx] = (_Float16)wrow[n];
        }
    }
    __syncthreads();

    const int wave = tid >> 5;
    const int lane = tid & 31;
    const long tileM = (long)blockIdx.x * 8 + wave;
    if (tileM * 16 >= (long)nrows) return;

    const long row0    = tileM * 16;
    const int  halfsel = lane >> 4;       // 0: lanes 0-15, 1: lanes 16-31
    const int  mlocal  = lane & 15;

    const float4* arow4 = reinterpret_cast<const float4*>(X + (row0 + mlocal) * DIN);
    const v16h*   bfrag = reinterpret_cast<const v16h*>(wlds);

    v8f acc0 = {}, acc1 = {}, acc2 = {}, acc3 = {};

    #pragma unroll
    for (int kt = 0; kt < 8; ++kt) {
        // A fragment: two groups of 8 consecutive f32, convert to f16
        const int g = kt * 8 + halfsel * 2;         // float4 index of group0
        float4 f0 = arow4[g + 0];
        float4 f1 = arow4[g + 1];
        float4 f2 = arow4[g + 4];
        float4 f3 = arow4[g + 5];
        v16h a;
        a[0]  = (_Float16)f0.x; a[1]  = (_Float16)f0.y;
        a[2]  = (_Float16)f0.z; a[3]  = (_Float16)f0.w;
        a[4]  = (_Float16)f1.x; a[5]  = (_Float16)f1.y;
        a[6]  = (_Float16)f1.z; a[7]  = (_Float16)f1.w;
        a[8]  = (_Float16)f2.x; a[9]  = (_Float16)f2.y;
        a[10] = (_Float16)f2.z; a[11] = (_Float16)f2.w;
        a[12] = (_Float16)f3.x; a[13] = (_Float16)f3.y;
        a[14] = (_Float16)f3.z; a[15] = (_Float16)f3.w;

        const v16h b0 = bfrag[(kt * 4 + 0) * 32 + lane];
        const v16h b1 = bfrag[(kt * 4 + 1) * 32 + lane];
        const v16h b2 = bfrag[(kt * 4 + 2) * 32 + lane];
        const v16h b3 = bfrag[(kt * 4 + 3) * 32 + lane];

        acc0 = __builtin_amdgcn_wmma_f32_16x16x32_f16(false, a, false, b0, (short)0, acc0, false, false);
        acc1 = __builtin_amdgcn_wmma_f32_16x16x32_f16(false, a, false, b1, (short)0, acc1, false, false);
        acc2 = __builtin_amdgcn_wmma_f32_16x16x32_f16(false, a, false, b2, (short)0, acc2, false, false);
        acc3 = __builtin_amdgcn_wmma_f32_16x16x32_f16(false, a, false, b3, (short)0, acc3, false, false);
    }

    // D layout: lanes 0-15 -> col=lane, row=r; lanes 16-31 -> col=lane-16, row=r+8
    const long rowbase = row0 + halfsel * 8;
    #pragma unroll
    for (int r = 0; r < 8; ++r) {
        float* orow = OUT + (rowbase + r) * DOUT + mlocal;
        orow[0]  = acc0[r];
        orow[16] = acc1[r];
        orow[32] = acc2[r];
        orow[48] = acc3[r];
    }
}

// ---------------------------------------------------------------------------
__global__ void zero_f4(float4* __restrict__ p, long n4)
{
    long i = (long)blockIdx.x * blockDim.x + threadIdx.x;
    if (i < n4) p[i] = make_float4(0.f, 0.f, 0.f, 0.f);
}

// ---------------------------------------------------------------------------
// SpMM scatter: OUT[row[e], :] += val[e] * T[col[e], :].  One wave per edge,
// 2 floats per lane. T and OUT are 25.6 MB each -> L2 resident.
// ---------------------------------------------------------------------------
__global__ __launch_bounds__(256) void spmm_scatter(
    const int* __restrict__ row, const int* __restrict__ col,
    const float* __restrict__ val, const float* __restrict__ T,
    float* __restrict__ OUT, int nedges)
{
    const long gid = (long)blockIdx.x * blockDim.x + threadIdx.x;
    const long e   = gid >> 5;
    if (e >= (long)nedges) return;
    const int lane = threadIdx.x & 31;

    const int   r = row[e];
    const int   c = col[e];
    const float v = val[e];

    const float2 t = *reinterpret_cast<const float2*>(T + (long)c * DOUT + lane * 2);
    float* dst = OUT + (long)r * DOUT + lane * 2;
    __hip_atomic_fetch_add(dst + 0, v * t.x, __ATOMIC_RELAXED, __HIP_MEMORY_SCOPE_AGENT);
    __hip_atomic_fetch_add(dst + 1, v * t.y, __ATOMIC_RELAXED, __HIP_MEMORY_SCOPE_AGENT);
}

// ---------------------------------------------------------------------------
// Fold attention matrices: vq[d] = sum_a w_query[d,a]*w_att[64+a],
//                          vk[d] = sum_a w_keys[d,a]*w_att[a].   64 threads.
// ---------------------------------------------------------------------------
__global__ void attvec(const float* __restrict__ wq, const float* __restrict__ wk,
                       const float* __restrict__ watt,
                       float* __restrict__ vq, float* __restrict__ vk)
{
    const int d = threadIdx.x;
    if (d >= DOUT) return;
    float sq = 0.f, sk = 0.f;
    #pragma unroll 8
    for (int a = 0; a < 64; ++a) {
        sq += wq[d * 64 + a] * watt[64 + a];
        sk += wk[d * 64 + a] * watt[a];
    }
    vq[d] = sq;
    vk[d] = sk;
}

// ---------------------------------------------------------------------------
// Per-node finalize: e0 = elu(self.(vk+vq)), e1 = elu(nb.vk + self.vq),
// att = softmax([e0,e1]), out = att0*self + att1*nb + bias.
// One wave per node, 2 floats/lane, shfl-xor wave32 reduction.
// ---------------------------------------------------------------------------
__global__ __launch_bounds__(256) void finalize(
    const float* __restrict__ selfft, const float* __restrict__ nbft,
    const float* __restrict__ vq, const float* __restrict__ vk,
    const float* __restrict__ bias, float* __restrict__ out,
    float* __restrict__ att_out, int nnodes)
{
    const long gid = (long)blockIdx.x * blockDim.x + threadIdx.x;
    const long n   = gid >> 5;
    if (n >= (long)nnodes) return;
    const int lane = threadIdx.x & 31;
    const int j    = lane * 2;

    const float2 s = *reinterpret_cast<const float2*>(selfft + n * DOUT + j);
    const float2 b = *reinterpret_cast<const float2*>(nbft  + n * DOUT + j);
    const float vk0 = vk[j], vk1 = vk[j + 1];
    const float vq0 = vq[j], vq1 = vq[j + 1];

    float p0 = s.x * (vk0 + vq0) + s.y * (vk1 + vq1);
    float p1 = b.x * vk0 + b.y * vk1 + s.x * vq0 + s.y * vq1;

    #pragma unroll
    for (int off = 16; off >= 1; off >>= 1) {
        p0 += __shfl_xor(p0, off, 32);
        p1 += __shfl_xor(p1, off, 32);
    }

    const float e0 = (p0 > 0.f) ? p0 : (expf(p0) - 1.f);
    const float e1 = (p1 > 0.f) ? p1 : (expf(p1) - 1.f);
    const float m  = fmaxf(e0, e1);
    const float x0 = expf(e0 - m);
    const float x1 = expf(e1 - m);
    const float inv = 1.f / (x0 + x1);
    const float a0 = x0 * inv;
    const float a1 = x1 * inv;

    const float2 bi = *reinterpret_cast<const float2*>(bias + j);
    float2 o;
    o.x = a0 * s.x + a1 * b.x + bi.x;
    o.y = a0 * s.y + a1 * b.y + bi.y;
    *reinterpret_cast<float2*>(out + n * DOUT + j) = o;

    if (lane == 0) {
        att_out[n * 2 + 0] = a0;
        att_out[n * 2 + 1] = a1;
    }
}

// ---------------------------------------------------------------------------
extern "C" void kernel_launch(void* const* d_in, const int* in_sizes, int n_in,
                              void* d_out, int out_size, void* d_ws, size_t ws_size,
                              hipStream_t stream)
{
    const float* x_p       = (const float*)d_in[0];
    const float* x_a       = (const float*)d_in[1];
    const int*   adj_a_row = (const int*)  d_in[2];
    const int*   adj_a_col = (const int*)  d_in[3];
    const float* adj_a_val = (const float*)d_in[4];
    const int*   adj_p_row = (const int*)  d_in[5];
    const int*   adj_p_col = (const int*)  d_in[6];
    const float* adj_p_val = (const float*)d_in[7];
    const float* w_self_p  = (const float*)d_in[8];
    const float* w_rel_p_a = (const float*)d_in[9];
    const float* bias_p    = (const float*)d_in[10];
    const float* w_query_p = (const float*)d_in[11];
    const float* w_keys_p  = (const float*)d_in[12];
    const float* w_att_p   = (const float*)d_in[13];
    const float* w_self_a  = (const float*)d_in[14];
    const float* w_rel_a_p = (const float*)d_in[15];
    const float* bias_a    = (const float*)d_in[16];
    const float* w_query_a = (const float*)d_in[17];
    const float* w_keys_a  = (const float*)d_in[18];
    const float* w_att_a   = (const float*)d_in[19];

    const int N = in_sizes[0] / DIN;        // 100000
    const int E = in_sizes[2];              // 3200000
    const long FT = (long)N * DOUT;         // 6.4M floats per feature buffer

    float* ws = (float*)d_ws;
    float* selfft_p = ws;
    float* selfft_a = ws + FT;
    float* T_a2p    = ws + 2 * FT;          // x_a @ w_rel_p_a
    float* T_p2a    = ws + 3 * FT;          // x_p @ w_rel_a_p
    float* nbft_p   = ws + 4 * FT;
    float* nbft_a   = ws + 5 * FT;
    float* vq_p     = ws + 6 * FT;
    float* vk_p     = vq_p + 64;
    float* vq_a     = vq_p + 128;
    float* vk_a     = vq_p + 192;

    float* out_p  = (float*)d_out;
    float* out_a  = out_p + FT;
    float* att_p  = out_p + 2 * FT;
    float* att_a  = att_p + (long)N * 2;

    // 1) dense GEMMs (WMMA)
    const int tiles      = (N + 15) / 16;
    const int gemmBlocks = (tiles + 7) / 8;
    gemm_wmma_256x64<<<gemmBlocks, 256, 0, stream>>>(x_p, w_self_p,  selfft_p, N);
    gemm_wmma_256x64<<<gemmBlocks, 256, 0, stream>>>(x_a, w_rel_p_a, T_a2p,    N);
    gemm_wmma_256x64<<<gemmBlocks, 256, 0, stream>>>(x_a, w_self_a,  selfft_a, N);
    gemm_wmma_256x64<<<gemmBlocks, 256, 0, stream>>>(x_p, w_rel_a_p, T_p2a,    N);

    // 2) zero both scatter buffers (contiguous: nbft_p then nbft_a)
    {
        const long n4 = (2 * FT) / 4;
        const int  zb = (int)((n4 + 255) / 256);
        zero_f4<<<zb, 256, 0, stream>>>((float4*)nbft_p, n4);
    }

    // 3) SpMM scatter-add (one wave per edge)
    {
        const int blocks = (E + 7) / 8;     // 8 waves (edges) per 256-thread block
        spmm_scatter<<<blocks, 256, 0, stream>>>(adj_a_row, adj_a_col, adj_a_val,
                                                 T_a2p, nbft_p, E);
        spmm_scatter<<<blocks, 256, 0, stream>>>(adj_p_row, adj_p_col, adj_p_val,
                                                 T_p2a, nbft_a, E);
    }

    // 4) fold attention matrices into 64-vectors
    attvec<<<1, 64, 0, stream>>>(w_query_p, w_keys_p, w_att_p, vq_p, vk_p);
    attvec<<<1, 64, 0, stream>>>(w_query_a, w_keys_a, w_att_a, vq_a, vk_a);

    // 5) attention + blend + bias
    {
        const long nthreads = (long)N * 32;
        const int  blocks   = (int)((nthreads + 255) / 256);
        finalize<<<blocks, 256, 0, stream>>>(selfft_p, nbft_p, vq_p, vk_p,
                                             bias_p, out_p, att_p, N);
        finalize<<<blocks, 256, 0, stream>>>(selfft_a, nbft_a, vq_a, vk_a,
                                             bias_a, out_a, att_a, N);
    }
}
